// Deep_Augmented_MUSIC_39264591020388
// MI455X (gfx1250) — matrix operand order repaired and checked by reading, not compiled, not run on hardware
//
#include <hip/hip_runtime.h>

// ---------------------------------------------------------------------------
// Deep Augmented MUSIC pipeline for gfx1250 (MI455X), wave32 + WMMA f16.
//   K-A: BN stats per timestep  -> s[t], o[t]
//   K-B: 512-step GRU (WMMA 16x16x32 f16, weights resident in VGPRs) + FC -> Rx
//   K-C: complex 16x16 eig (Givens Hessenberg + shifted QR), noise subspace Un
//   K-D: MUSIC spectrum (360 angles)
//   K-E: small MLP -> (1024,3)
// ---------------------------------------------------------------------------

typedef __attribute__((ext_vector_type(16))) _Float16 v16h;
typedef __attribute__((ext_vector_type(8)))  float    v8f;

union V16H { v16h v; _Float16 e[16]; };

#define N_ANT   16
#define T_SNAP  512
#define M_SRC   3
#define BATCH   1024
#define N_ANG   360
#define HID     32   // 2*N_ANT

__device__ __forceinline__ float sigm(float x) { return 1.f / (1.f + __expf(-x)); }

// ------------------------------ complex helpers ----------------------------
__device__ __forceinline__ float2 cmul(float2 a, float2 b) {
    return make_float2(a.x*b.x - a.y*b.y, a.x*b.y + a.y*b.x);
}
__device__ __forceinline__ float2 conjf2(float2 a) { return make_float2(a.x, -a.y); }
__device__ __forceinline__ float2 cadd(float2 a, float2 b) { return make_float2(a.x+b.x, a.y+b.y); }
__device__ __forceinline__ float2 csub(float2 a, float2 b) { return make_float2(a.x-b.x, a.y-b.y); }
__device__ __forceinline__ float2 cscale(float2 a, float s){ return make_float2(a.x*s, a.y*s); }
__device__ __forceinline__ float  cabs2(float2 a) { return a.x*a.x + a.y*a.y; }
__device__ __forceinline__ float2 csqrtf2(float2 w) {
    float r  = sqrtf(w.x*w.x + w.y*w.y);
    float re = sqrtf(fmaxf(0.f, 0.5f*(r + w.x)));
    float im = sqrtf(fmaxf(0.f, 0.5f*(r - w.x)));
    return make_float2(re, (w.y < 0.f) ? -im : im);
}
__device__ __forceinline__ void givens(float2 a, float2 b, float2& ca, float2& sb) {
    float r2 = a.x*a.x + a.y*a.y + b.x*b.x + b.y*b.y;
    if (r2 < 1e-36f) { ca = make_float2(1.f, 0.f); sb = make_float2(0.f, 0.f); return; }
    float inv = rsqrtf(r2);
    ca = make_float2(a.x*inv, -a.y*inv);   // conj(a)/r
    sb = make_float2(b.x*inv, -b.y*inv);   // conj(b)/r
}
#define WB() __builtin_amdgcn_wave_barrier()

// ---------------------------------------------------------------------------
// Kernel A: per-timestep mean/var over (batch, 32 features) -> scale/offset
// x[b,t,c] = S[b*8192 + (nant&15)*512 + (t&15)*32 + c], S = Xr or Xi by t>=256
// ---------------------------------------------------------------------------
__global__ __launch_bounds__(256)
void bn_stats_kernel(const float* __restrict__ Xr, const float* __restrict__ Xi,
                     const float* __restrict__ bnw, const float* __restrict__ bnb,
                     float* __restrict__ s_arr, float* __restrict__ o_arr)
{
    const int t    = blockIdx.x;
    const int nant = t >> 4;
    const float* src  = (nant < 16) ? Xr : Xi;
    const float* base = src + (nant & 15) * 512 + (t & 15) * 32;
    const int tid = threadIdx.x;

    float s = 0.f, s2 = 0.f;
    for (int i = tid; i < BATCH * 32; i += 256) {
        int b = i >> 5, c = i & 31;
        float x = base[(size_t)b * 8192 + c];
        s += x; s2 += x * x;
    }
    __shared__ float rs[256], rq[256];
    rs[tid] = s; rq[tid] = s2;
    __syncthreads();
    for (int off = 128; off > 0; off >>= 1) {
        if (tid < off) { rs[tid] += rs[tid + off]; rq[tid] += rq[tid + off]; }
        __syncthreads();
    }
    if (tid == 0) {
        const float invN = 1.f / 32768.f;
        float mean = rs[0] * invN;
        float var  = rq[0] * invN - mean * mean;
        float inv  = rsqrtf(var + 1e-5f);
        float sc   = inv * bnw[t];
        s_arr[t] = sc;
        o_arr[t] = bnb[t] - mean * sc;
    }
}

// ---------------------------------------------------------------------------
// Kernel B: GRU over 512 steps + final FC (h @ fc_W.T + fc_b) -> Rx (B,512)
// One wave per 16-batch tile; 4 waves/block, 16 blocks.
// WMMA f16 16x16x32: A = x_t or h (16x32), B = W.T tiles (32x16), C f32.
// ---------------------------------------------------------------------------
__global__ __launch_bounds__(128)
void gru_fc_kernel(const float* __restrict__ Xr, const float* __restrict__ Xi,
                   const float* __restrict__ Wih, const float* __restrict__ Whh,
                   const float* __restrict__ bih, const float* __restrict__ bhh,
                   const float* __restrict__ fcW, const float* __restrict__ fcb,
                   const float* __restrict__ s_arr, const float* __restrict__ o_arr,
                   float* __restrict__ Rx)
{
    const int lane = threadIdx.x & 31;
    const int wv   = threadIdx.x >> 5;
    const int b0   = (blockIdx.x * 4 + wv) * 16;  // batch tile base
    const int m    = lane & 15;                   // row (batch within tile) / B col
    const int kh   = lane >> 4;                   // lane half (K split)

    __shared__ _Float16 hbuf[4][16 * 32];
    _Float16* hl = hbuf[wv];
    for (int i = lane; i < 512; i += 32) hl[i] = (_Float16)0.f;

    // Resident B fragments: 6 gate tiles each for Wih, Whh (f32 -> f16).
    // B layout (16-bit, 32x16): lane n = L%16, bfrag[i] = W[16*tile+n][16*kh + i]
    V16H wiF[6], whF[6];
    float bi[6], bh[6];
    #pragma unroll
    for (int tt = 0; tt < 6; ++tt) {
        const int row = tt * 16 + m;
        const float* wr = Wih + row * HID + kh * 16;
        const float* hr = Whh + row * HID + kh * 16;
        #pragma unroll
        for (int i = 0; i < 16; ++i) {
            wiF[tt].e[i] = (_Float16)wr[i];
            whF[tt].e[i] = (_Float16)hr[i];
        }
        bi[tt] = bih[row];
        bh[tt] = bhh[row];
    }

    v8f hreg[2] = {};   // h state in C-layout (f32), cols [0,16) and [16,32)

    for (int t = 0; t < T_SNAP; ++t) {
        WB();
        // ---- A fragment of x_t, normalized, straight from global ----
        const int nant = t >> 4;
        const float* src = (nant < 16) ? Xr : Xi;
        const float sc = s_arr[t];
        const float of = o_arr[t];
        const float* xb = src + (size_t)(b0 + m) * 8192 + (nant & 15) * 512 + (t & 15) * 32;
        V16H xa;
        #pragma unroll
        for (int i = 0; i < 8; ++i) {
            xa.e[i]     = (_Float16)(xb[16 * kh + i]      * sc + of);
            xa.e[8 + i] = (_Float16)(xb[16 * kh + 8 + i] * sc + of);
        }
        // ---- A fragment of h from LDS ----
        V16H ha;
        #pragma unroll
        for (int i = 0; i < 8; ++i) {
            ha.e[i]     = hl[m * 32 + 16 * kh + i];
            ha.e[8 + i] = hl[m * 32 + 16 * kh + 8 + i];
        }
        // ---- 12 WMMAs: gi = x W_ih^T, gh = h W_hh^T ----
        v8f gi[6], gh[6];
        #pragma unroll
        for (int tt = 0; tt < 6; ++tt) {
            v8f z = {};
            gi[tt] = __builtin_amdgcn_wmma_f32_16x16x32_f16(false, xa.v, false, wiF[tt].v,
                                                            (short)0, z, false, false);
            gh[tt] = __builtin_amdgcn_wmma_f32_16x16x32_f16(false, ha.v, false, whF[tt].v,
                                                            (short)0, z, false, false);
        }
        // ---- gates + state update (tiles 0,1 = r | 2,3 = z | 4,5 = n) ----
        #pragma unroll
        for (int tc = 0; tc < 2; ++tc) {
            #pragma unroll
            for (int q = 0; q < 8; ++q) {
                float r  = sigm((gi[tc][q]     + bi[tc])     + (gh[tc][q]     + bh[tc]));
                float zz = sigm((gi[tc + 2][q] + bi[tc + 2]) + (gh[tc + 2][q] + bh[tc + 2]));
                float ng = tanhf((gi[tc + 4][q] + bi[tc + 4]) + r * (gh[tc + 4][q] + bh[tc + 4]));
                float hn = (1.f - zz) * ng + zz * hreg[tc][q];
                hreg[tc][q] = hn;
                hl[(q + 8 * kh) * 32 + tc * 16 + m] = (_Float16)hn;
            }
        }
    }

    WB();
    // ---- final FC: Rx = h @ fc_W.T + fc_b  (32 output tiles of 16) ----
    V16H ha;
    #pragma unroll
    for (int i = 0; i < 8; ++i) {
        ha.e[i]     = hl[m * 32 + 16 * kh + i];
        ha.e[8 + i] = hl[m * 32 + 16 * kh + 8 + i];
    }
    for (int T = 0; T < 32; ++T) {
        V16H bf;
        const float* wr = fcW + (size_t)(T * 16 + m) * HID + kh * 16;
        #pragma unroll
        for (int i = 0; i < 16; ++i) bf.e[i] = (_Float16)wr[i];
        v8f z = {};
        v8f acc = __builtin_amdgcn_wmma_f32_16x16x32_f16(false, ha.v, false, bf.v,
                                                         (short)0, z, false, false);
        const float bias = fcb[T * 16 + m];
        #pragma unroll
        for (int q = 0; q < 8; ++q)
            Rx[(size_t)(b0 + q + 8 * kh) * 512 + T * 16 + m] = acc[q] + bias;
    }
}

// ---------------------------------------------------------------------------
// Kernel C: complex 16x16 eig per batch. 16 lanes / matrix, 4 matrices/block.
// Givens Hessenberg -> Wilkinson-shifted QR (fixed schedule) -> back-subst
// eigenvectors -> |lambda|-descending select of columns 3..15 -> Un.
// ---------------------------------------------------------------------------
__global__ __launch_bounds__(64)
void eig_kernel(const float* __restrict__ Rx,
                float* __restrict__ unr, float* __restrict__ uni)
{
    const int g = threadIdx.x >> 4;
    const int l = threadIdx.x & 15;
    const int b = blockIdx.x * 4 + g;

    __shared__ volatile float Hr[4][16][17], Hi[4][16][17];
    __shared__ volatile float Qr[4][16][17], Qi[4][16][17];
    __shared__ volatile float Yr[4][16][16], Yi[4][16][16];
    __shared__ volatile float rcx[4][16], rcy[4][16], rsx[4][16], rsy[4][16];
    __shared__ volatile float absl[4][16];

#define LDH(i,j)   make_float2(Hr[g][(i)][(j)], Hi[g][(i)][(j)])
#define STH(i,j,v) do { float2 _t = (v); Hr[g][(i)][(j)] = _t.x; Hi[g][(i)][(j)] = _t.y; } while (0)
#define LDQ(i,j)   make_float2(Qr[g][(i)][(j)], Qi[g][(i)][(j)])
#define STQ(i,j,v) do { float2 _t = (v); Qr[g][(i)][(j)] = _t.x; Qi[g][(i)][(j)] = _t.y; } while (0)

    // load K = Rx[:16,:] + i*Rx[16:,:] ; Q = I
    const float* rb = Rx + (size_t)b * 512;
    for (int i = 0; i < 16; ++i) {
        Hr[g][i][l] = rb[i * 16 + l];
        Hi[g][i][l] = rb[256 + i * 16 + l];
        Qr[g][i][l] = (i == l) ? 1.f : 0.f;
        Qi[g][i][l] = 0.f;
    }
    WB();

    // ---- Hessenberg reduction by Givens similarity ----
    for (int j = 0; j < 14; ++j) {
        for (int i = 15; i >= j + 2; --i) {
            float2 a  = LDH(i - 1, j);
            float2 bb = LDH(i, j);
            float2 ca, sb; givens(a, bb, ca, sb);
            WB();
            // left: rows (i-1, i), lane l = column
            float2 t0 = LDH(i - 1, l), t1 = LDH(i, l);
            STH(i - 1, l, cadd(cmul(ca, t0), cmul(sb, t1)));
            STH(i,     l, csub(cmul(conjf2(ca), t1), cmul(conjf2(sb), t0)));
            WB();
            // right (similarity): cols (i-1, i), lane l = row; also Q
            float2 u = LDH(l, i - 1), w = LDH(l, i);
            STH(l, i - 1, cadd(cmul(u, conjf2(ca)), cmul(w, conjf2(sb))));
            STH(l, i,     csub(cmul(w, ca),         cmul(u, sb)));
            float2 qu = LDQ(l, i - 1), qw = LDQ(l, i);
            STQ(l, i - 1, cadd(cmul(qu, conjf2(ca)), cmul(qw, conjf2(sb))));
            STQ(l, i,     csub(cmul(qw, ca),         cmul(qu, sb)));
            WB();
        }
    }

    // ---- shifted QR, fixed deflation schedule ----
    for (int mm = 15; mm >= 1; --mm) {
        for (int it = 0; it < 6; ++it) {
            // Wilkinson shift from trailing 2x2 (computed redundantly by all lanes)
            float2 A_ = LDH(mm - 1, mm - 1), B_ = LDH(mm - 1, mm);
            float2 C_ = LDH(mm, mm - 1),     D_ = LDH(mm, mm);
            float2 mC = cscale(cadd(A_, D_), 0.5f);
            float2 hh = cscale(csub(A_, D_), 0.5f);
            float2 sq = csqrtf2(cadd(cmul(hh, hh), cmul(B_, C_)));
            float2 mu1 = cadd(mC, sq), mu2 = csub(mC, sq);
            float2 sig = (cabs2(csub(mu1, D_)) < cabs2(csub(mu2, D_))) ? mu1 : mu2;
            WB();
            if (l <= mm) { Hr[g][l][l] -= sig.x; Hi[g][l][l] -= sig.y; }
            WB();
            // left factorization
            for (int k = 0; k < mm; ++k) {
                float2 a  = LDH(k, k);
                float2 bb = LDH(k + 1, k);
                float2 ca, sb; givens(a, bb, ca, sb);
                if (l == 0) { rcx[g][k] = ca.x; rcy[g][k] = ca.y;
                              rsx[g][k] = sb.x; rsy[g][k] = sb.y; }
                WB();
                float2 t0 = LDH(k, l), t1 = LDH(k + 1, l);
                STH(k,     l, cadd(cmul(ca, t0), cmul(sb, t1)));
                STH(k + 1, l, csub(cmul(conjf2(ca), t1), cmul(conjf2(sb), t0)));
                WB();
            }
            // RQ + Q accumulation
            for (int k = 0; k < mm; ++k) {
                float2 ca = make_float2(rcx[g][k], rcy[g][k]);
                float2 sb = make_float2(rsx[g][k], rsy[g][k]);
                float2 u = LDH(l, k), w = LDH(l, k + 1);
                STH(l, k,     cadd(cmul(u, conjf2(ca)), cmul(w, conjf2(sb))));
                STH(l, k + 1, csub(cmul(w, ca),         cmul(u, sb)));
                float2 qu = LDQ(l, k), qw = LDQ(l, k + 1);
                STQ(l, k,     cadd(cmul(qu, conjf2(ca)), cmul(qw, conjf2(sb))));
                STQ(l, k + 1, csub(cmul(qw, ca),         cmul(qu, sb)));
                WB();
            }
            if (l <= mm) { Hr[g][l][l] += sig.x; Hi[g][l][l] += sig.y; }
            WB();
        }
        if (l == 0) { Hr[g][mm][mm - 1] = 0.f; Hi[g][mm][mm - 1] = 0.f; }
        WB();
    }

    // ---- eigenvectors of (upper-triangular) T by back-substitution, lane l owns λ_l ----
    float2 lam = LDH(l, l);
    for (int i = 0; i < 16; ++i) { Yr[g][i][l] = 0.f; Yi[g][i][l] = 0.f; }
    Yr[g][l][l] = 1.f;
    for (int k = l - 1; k >= 0; --k) {
        float2 num = make_float2(0.f, 0.f);
        for (int i = k + 1; i <= l; ++i) {
            float2 yi = make_float2(Yr[g][i][l], Yi[g][i][l]);
            num = cadd(num, cmul(LDH(k, i), yi));
        }
        float2 den = csub(LDH(k, k), lam);
        float d2 = cabs2(den);
        if (d2 < 1e-24f) { den = make_float2(1e-12f, 0.f); d2 = 1e-24f; }
        float2 yk = cscale(cmul(num, conjf2(den)), -1.f / d2);  // -num/den
        Yr[g][k][l] = yk.x; Yi[g][k][l] = yk.y;
    }
    WB();
    // v = Q y ; normalize
    float2 vv[16];
    float nrm = 0.f;
    #pragma unroll
    for (int i = 0; i < 16; ++i) {
        float2 acc = make_float2(0.f, 0.f);
        #pragma unroll
        for (int k = 0; k < 16; ++k) {
            float2 y = make_float2(Yr[g][k][l], Yi[g][k][l]);
            acc = cadd(acc, cmul(LDQ(i, k), y));
        }
        vv[i] = acc; nrm += cabs2(acc);
    }
    float inv = rsqrtf(fmaxf(nrm, 1e-30f));

    // rank by |lambda| descending (flip of stable ascending argsort)
    absl[g][l] = sqrtf(cabs2(lam));
    WB();
    float myA = absl[g][l];
    int rank = 0;
    for (int i = 0; i < 16; ++i) {
        float ai = absl[g][i];
        if (ai > myA || (ai == myA && i > l)) rank++;
    }
    if (rank >= M_SRC) {
        int kk = rank - M_SRC;                     // 0..12
        float* ur = unr + (size_t)b * 208;
        float* ui = uni + (size_t)b * 208;
        #pragma unroll
        for (int i = 0; i < 16; ++i) {
            ur[i * 13 + kk] = vv[i].x * inv;
            ui[i * 13 + kk] = vv[i].y * inv;
        }
    }
#undef LDH
#undef STH
#undef LDQ
#undef STQ
}

// ---------------------------------------------------------------------------
// Kernel D: MUSIC spectrum. One block per batch, thread = angle.
// spec[b,a] = 1 / sum_k | sum_n exp(+i*pi*n*sin(th_a)) * Un[n,k] |^2
// ---------------------------------------------------------------------------
__global__ __launch_bounds__(384)
void spectrum_kernel(const float* __restrict__ unr, const float* __restrict__ uni,
                     float* __restrict__ spec)
{
    const int b = blockIdx.x;
    const int a = threadIdx.x;
    __shared__ float Ur[16][13], Ui[16][13];
    if (threadIdx.x < 208) {
        ((float*)Ur)[threadIdx.x] = unr[(size_t)b * 208 + threadIdx.x];
        ((float*)Ui)[threadIdx.x] = uni[(size_t)b * 208 + threadIdx.x];
    }
    __syncthreads();

    if (a < N_ANG) {
        const float PI = 3.14159265358979f;
        float theta = -0.5f * PI + (float)a * (PI / 359.f);
        float st = __sinf(theta);
        float sr[13], si[13];
        #pragma unroll
        for (int k = 0; k < 13; ++k) { sr[k] = 0.f; si[k] = 0.f; }
        for (int n = 0; n < 16; ++n) {
            float ph = PI * (float)n * st;
            float sn, cn;
            __sincosf(ph, &sn, &cn);          // conj(A) = cn + i*sn
            #pragma unroll
            for (int k = 0; k < 13; ++k) {
                float urr = Ur[n][k], uii = Ui[n][k];
                sr[k] += cn * urr - sn * uii;
                si[k] += cn * uii + sn * urr;
            }
        }
        float den = 0.f;
        #pragma unroll
        for (int k = 0; k < 13; ++k) den += sr[k] * sr[k] + si[k] * si[k];
        spec[(size_t)b * 384 + a] = 1.f / fmaxf(den, 1e-30f);
    } else {
        spec[(size_t)b * 384 + a] = 0.f;
    }
}

// ---------------------------------------------------------------------------
// Kernel E: MLP head. One thread per batch element (weights uniform -> s_loads).
// ---------------------------------------------------------------------------
__global__ __launch_bounds__(256)
void mlp_kernel(const float* __restrict__ spec,
                const float* __restrict__ fc1W, const float* __restrict__ fc1b,
                const float* __restrict__ fc2W, const float* __restrict__ fc2b,
                const float* __restrict__ fc3W, const float* __restrict__ fc3b,
                float* __restrict__ out)
{
    const int b = blockIdx.x * 256 + threadIdx.x;
    const float* sp = spec + (size_t)b * 384;

    float y1[32];
    #pragma unroll
    for (int o = 0; o < 32; ++o) y1[o] = fc1b[o];
    for (int a = 0; a < N_ANG; ++a) {
        float sv = sp[a];
        #pragma unroll
        for (int o = 0; o < 32; ++o) y1[o] = fmaf(sv, fc1W[o * 360 + a], y1[o]);
    }
    #pragma unroll
    for (int o = 0; o < 32; ++o) y1[o] = fmaxf(y1[o], 0.f);

    float y2[32];
    #pragma unroll
    for (int o = 0; o < 32; ++o) {
        float acc = fc2b[o];
        #pragma unroll
        for (int k = 0; k < 32; ++k) acc = fmaf(fc2W[o * 32 + k], y1[k], acc);
        y2[o] = fmaxf(acc, 0.f);
    }
    float y3[32];
    #pragma unroll
    for (int o = 0; o < 32; ++o) {
        float acc = fc2b[o];
        #pragma unroll
        for (int k = 0; k < 32; ++k) acc = fmaf(fc2W[o * 32 + k], y2[k], acc);
        y3[o] = fmaxf(acc, 0.f);
    }
    #pragma unroll
    for (int j = 0; j < M_SRC; ++j) {
        float acc = fc3b[j];
        #pragma unroll
        for (int k = 0; k < 32; ++k) acc = fmaf(fc3W[j * 32 + k], y3[k], acc);
        out[(size_t)b * M_SRC + j] = acc;
    }
}

// ---------------------------------------------------------------------------
extern "C" void kernel_launch(void* const* d_in, const int* in_sizes, int n_in,
                              void* d_out, int out_size, void* d_ws, size_t ws_size,
                              hipStream_t stream)
{
    (void)in_sizes; (void)n_in; (void)out_size; (void)ws_size;
    const float* Xr   = (const float*)d_in[0];
    const float* Xi   = (const float*)d_in[1];
    const float* bnw  = (const float*)d_in[2];
    const float* bnb  = (const float*)d_in[3];
    const float* Wih  = (const float*)d_in[4];
    const float* Whh  = (const float*)d_in[5];
    const float* bih  = (const float*)d_in[6];
    const float* bhh  = (const float*)d_in[7];
    const float* fcW  = (const float*)d_in[8];
    const float* fcb  = (const float*)d_in[9];
    const float* fc1W = (const float*)d_in[10];
    const float* fc1b = (const float*)d_in[11];
    const float* fc2W = (const float*)d_in[12];
    const float* fc2b = (const float*)d_in[13];
    const float* fc3W = (const float*)d_in[14];
    const float* fc3b = (const float*)d_in[15];

    float* ws    = (float*)d_ws;
    float* s_arr = ws;                       // 512
    float* o_arr = ws + 512;                 // 512
    float* Rx    = ws + 1024;                // 1024*512
    float* unr   = Rx + (size_t)BATCH * 512; // 1024*208
    float* uni   = unr + (size_t)BATCH * 208;
    float* spec  = uni + (size_t)BATCH * 208; // 1024*384

    bn_stats_kernel<<<T_SNAP, 256, 0, stream>>>(Xr, Xi, bnw, bnb, s_arr, o_arr);
    gru_fc_kernel<<<16, 128, 0, stream>>>(Xr, Xi, Wih, Whh, bih, bhh,
                                          fcW, fcb, s_arr, o_arr, Rx);
    eig_kernel<<<BATCH / 4, 64, 0, stream>>>(Rx, unr, uni);
    spectrum_kernel<<<BATCH, 384, 0, stream>>>(unr, uni, spec);
    mlp_kernel<<<BATCH / 256, 256, 0, stream>>>(spec, fc1W, fc1b, fc2W, fc2b,
                                                fc3W, fc3b, (float*)d_out);
}
